// PowerAttention_38680475468388
// MI455X (gfx1250) — matrix operand order repaired and checked by reading, not compiled, not run on hardware
//
#include <hip/hip_runtime.h>
#include <hip/hip_bf16.h>

// ---------------------------------------------------------------------------
// CDNA5 (gfx1250) power-attention pipeline, bf16 WMMA everywhere.
//   H=8 kv heads, R=2 -> HQ=16 query heads, D=128, DEG=2, T=C=2048.
// Round 2: register-blocked GEMM (32x64 per wave, 8 wmma / 12 b128 per K-step)
// ---------------------------------------------------------------------------

typedef __attribute__((ext_vector_type(16))) __bf16 bf16x16;
typedef __attribute__((ext_vector_type(8)))  float  f32x8;

#define T_DIM   2048
#define C_DIM   2048
#define QDIM    2048
#define KVDIM   1024
#define NHEADS  8
#define HQ      16
#define HEADD   128
#define GATE_OFFSET 6.906768f

union BPair { unsigned int u; __bf16 b[2]; unsigned short s[2]; };

static __device__ __forceinline__ unsigned short f32_to_bf16_rne(float f) {
  unsigned int u = __float_as_uint(f);
  unsigned int r = u + 0x7FFFu + ((u >> 16) & 1u);
  return (unsigned short)(r >> 16);
}

static __device__ __forceinline__ void put_pair(bf16x16& v, int i, unsigned int u) {
  BPair p; p.u = u;
  v[2 * i]     = p.b[0];
  v[2 * i + 1] = p.b[1];
}

// B-operand fragment: lane holds column n = lane&15, K = 16*(lane>>4)+j,
// j = 0..15 contiguous in memory -> two 128-bit loads.
static __device__ __forceinline__ bf16x16 load_row16(const unsigned short* p) {
  const uint4* q = (const uint4*)p;
  uint4 w0 = q[0];
  uint4 w1 = q[1];
  bf16x16 v;
  put_pair(v, 0, w0.x); put_pair(v, 1, w0.y); put_pair(v, 2, w0.z); put_pair(v, 3, w0.w);
  put_pair(v, 4, w1.x); put_pair(v, 5, w1.y); put_pair(v, 6, w1.z); put_pair(v, 7, w1.w);
  return v;
}

// A-operand fragment: lane holds row m = lane&15; element pair i sits at
// K0 = 16*(lane>>4) + 2*i  (contiguous order, matching the B fragments).
static __device__ __forceinline__ bf16x16 load_a_frag(const unsigned short* rowp, int h) {
  bf16x16 v;
#pragma unroll
  for (int i = 0; i < 8; ++i) {
    int k0 = 16 * h + 2 * i;
    unsigned int u = *(const unsigned int*)(rowp + k0);
    put_pair(v, i, u);
  }
  return v;
}

static __device__ __forceinline__ f32x8 wmma_bf16(bf16x16 a, bf16x16 b, f32x8 c) {
  return __builtin_amdgcn_wmma_f32_16x16x32_bf16(false, a, false, b, (short)0, c,
                                                 false, false);
}

// ---------------------------------------------------------------------------
// Elementwise conversions
// ---------------------------------------------------------------------------
__global__ void conv_f32_to_bf16(const float* __restrict__ in,
                                 unsigned short* __restrict__ out, int n) {
  int i = blockIdx.x * blockDim.x + threadIdx.x;
  if (i < n) out[i] = f32_to_bf16_rne(in[i]);
}

// W: [K][N] row-major f32  ->  WT: [N][K] bf16
__global__ void conv_transpose_f32_to_bf16(const float* __restrict__ W,
                                           unsigned short* __restrict__ WT,
                                           int K, int N) {
  int i = blockIdx.x * blockDim.x + threadIdx.x;
  if (i < K * N) {
    int n = i / K;
    int k = i - n * K;
    WT[i] = f32_to_bf16_rne(W[(size_t)k * N + n]);
  }
}

// ---------------------------------------------------------------------------
// Gate: g = X @ Wg (N=8), log-sigmoid, then cumsum over T
// ---------------------------------------------------------------------------
__global__ void __launch_bounds__(256)
gate_logsig(const float* __restrict__ X, const float* __restrict__ Wg,
            float* __restrict__ lg) {
  int t    = blockIdx.x;
  int lane = threadIdx.x & 31;
  int hidx = threadIdx.x >> 5;   // 8 waves -> 8 kv heads
  const float* xr = X + (size_t)t * C_DIM;
  float s = 0.f;
  for (int c = lane; c < C_DIM; c += 32) s += xr[c] * Wg[(size_t)c * NHEADS + hidx];
  s += __shfl_xor(s, 16); s += __shfl_xor(s, 8);
  s += __shfl_xor(s, 4);  s += __shfl_xor(s, 2); s += __shfl_xor(s, 1);
  if (lane == 0) {
    float x = GATE_OFFSET + s;
    lg[t * NHEADS + hidx] = -log1pf(__expf(-x));   // log_sigmoid(x)
  }
}

__global__ void gate_cumsum(const float* __restrict__ lg, float* __restrict__ lG) {
  int hidx = threadIdx.x;
  if (hidx >= NHEADS) return;
  float acc = 0.f;
  for (int t = 0; t < T_DIM; ++t) {
    acc += lg[t * NHEADS + hidx];
    lG[t * NHEADS + hidx] = acc;
  }
}

// ---------------------------------------------------------------------------
// GEMM: C[M,N] = A[M,K] * B[K,N], A row-major bf16, B given transposed
// (BT[N][K] bf16).  Register-blocked: each wave computes a 32x64 tile
// (2 A-frags x 4 B-frags -> 8 WMMA per 32-deep K step, 12 b128 loads).
// Block = 8 waves arranged 4(M) x 2(N) -> 128x128 block tile.
// M, N must be multiples of 128 (true for all call sites).
// ---------------------------------------------------------------------------
__global__ void __launch_bounds__(256)
gemm_bf16_bf16out(const unsigned short* __restrict__ A,
                  const unsigned short* __restrict__ BT,
                  unsigned short* __restrict__ Cb,
                  int K, int lda, int ldbt, int ldc, int transOut) {
  const int lane = threadIdx.x & 31;
  const int wave = threadIdx.x >> 5;
  const int h    = lane >> 4;
  const int nn   = lane & 15;
  const int m0 = blockIdx.y * 128 + (wave >> 1) * 32;
  const int n0 = blockIdx.x * 128 + (wave & 1) * 64;

  const unsigned short* arow0 = A + (size_t)(m0 + nn) * lda;
  const unsigned short* arow1 = A + (size_t)(m0 + 16 + nn) * lda;
  const unsigned short* brow0 = BT + (size_t)(n0 + nn)      * ldbt + 16 * h;
  const unsigned short* brow1 = BT + (size_t)(n0 + 16 + nn) * ldbt + 16 * h;
  const unsigned short* brow2 = BT + (size_t)(n0 + 32 + nn) * ldbt + 16 * h;
  const unsigned short* brow3 = BT + (size_t)(n0 + 48 + nn) * ldbt + 16 * h;

  f32x8 acc[2][4];
#pragma unroll
  for (int i = 0; i < 2; ++i)
#pragma unroll
    for (int j = 0; j < 4; ++j) acc[i][j] = (f32x8){};

  for (int k0 = 0; k0 < K; k0 += 32) {
    __builtin_prefetch(arow0 + k0 + 128, 0, 1);
    __builtin_prefetch(brow0 + k0 + 128, 0, 1);
    bf16x16 a0 = load_a_frag(arow0 + k0, h);
    bf16x16 a1 = load_a_frag(arow1 + k0, h);
    bf16x16 b0 = load_row16(brow0 + k0);
    bf16x16 b1 = load_row16(brow1 + k0);
    bf16x16 b2 = load_row16(brow2 + k0);
    bf16x16 b3 = load_row16(brow3 + k0);
    acc[0][0] = wmma_bf16(a0, b0, acc[0][0]);
    acc[1][0] = wmma_bf16(a1, b0, acc[1][0]);
    acc[0][1] = wmma_bf16(a0, b1, acc[0][1]);
    acc[1][1] = wmma_bf16(a1, b1, acc[1][1]);
    acc[0][2] = wmma_bf16(a0, b2, acc[0][2]);
    acc[1][2] = wmma_bf16(a1, b2, acc[1][2]);
    acc[0][3] = wmma_bf16(a0, b3, acc[0][3]);
    acc[1][3] = wmma_bf16(a1, b3, acc[1][3]);
  }

#pragma unroll
  for (int i = 0; i < 2; ++i) {
#pragma unroll
    for (int j = 0; j < 4; ++j) {
#pragma unroll
      for (int r = 0; r < 8; ++r) {
        int m = m0 + 16 * i + r + 8 * h;   // C/D layout: VGPR r -> row r + 8*half
        int n = n0 + 16 * j + nn;
        unsigned short val = f32_to_bf16_rne(acc[i][j][r]);
        if (transOut) Cb[(size_t)n * ldc + m] = val;
        else          Cb[(size_t)m * ldc + n] = val;
      }
    }
  }
}

__global__ void __launch_bounds__(256)
gemm_bf16_f32out(const unsigned short* __restrict__ A,
                 const unsigned short* __restrict__ BT,
                 const float* __restrict__ bias, float* __restrict__ C,
                 int K, int lda, int ldbt, int ldc) {
  const int lane = threadIdx.x & 31;
  const int wave = threadIdx.x >> 5;
  const int h    = lane >> 4;
  const int nn   = lane & 15;
  const int m0 = blockIdx.y * 128 + (wave >> 1) * 32;
  const int n0 = blockIdx.x * 128 + (wave & 1) * 64;

  const unsigned short* arow0 = A + (size_t)(m0 + nn) * lda;
  const unsigned short* arow1 = A + (size_t)(m0 + 16 + nn) * lda;
  const unsigned short* brow0 = BT + (size_t)(n0 + nn)      * ldbt + 16 * h;
  const unsigned short* brow1 = BT + (size_t)(n0 + 16 + nn) * ldbt + 16 * h;
  const unsigned short* brow2 = BT + (size_t)(n0 + 32 + nn) * ldbt + 16 * h;
  const unsigned short* brow3 = BT + (size_t)(n0 + 48 + nn) * ldbt + 16 * h;

  f32x8 acc[2][4];
#pragma unroll
  for (int i = 0; i < 2; ++i)
#pragma unroll
    for (int j = 0; j < 4; ++j) acc[i][j] = (f32x8){};

  for (int k0 = 0; k0 < K; k0 += 32) {
    __builtin_prefetch(arow0 + k0 + 128, 0, 1);
    __builtin_prefetch(brow0 + k0 + 128, 0, 1);
    bf16x16 a0 = load_a_frag(arow0 + k0, h);
    bf16x16 a1 = load_a_frag(arow1 + k0, h);
    bf16x16 b0 = load_row16(brow0 + k0);
    bf16x16 b1 = load_row16(brow1 + k0);
    bf16x16 b2 = load_row16(brow2 + k0);
    bf16x16 b3 = load_row16(brow3 + k0);
    acc[0][0] = wmma_bf16(a0, b0, acc[0][0]);
    acc[1][0] = wmma_bf16(a1, b0, acc[1][0]);
    acc[0][1] = wmma_bf16(a0, b1, acc[0][1]);
    acc[1][1] = wmma_bf16(a1, b1, acc[1][1]);
    acc[0][2] = wmma_bf16(a0, b2, acc[0][2]);
    acc[1][2] = wmma_bf16(a1, b2, acc[1][2]);
    acc[0][3] = wmma_bf16(a0, b3, acc[0][3]);
    acc[1][3] = wmma_bf16(a1, b3, acc[1][3]);
  }

#pragma unroll
  for (int i = 0; i < 2; ++i) {
#pragma unroll
    for (int j = 0; j < 4; ++j) {
      float b = bias[n0 + 16 * j + nn];
#pragma unroll
      for (int r = 0; r < 8; ++r) {
        int m = m0 + 16 * i + r + 8 * h;
        C[(size_t)m * ldc + n0 + 16 * j + nn] = acc[i][j][r] + b;
      }
    }
  }
}

// ---------------------------------------------------------------------------
// Power attention: one wave per (16-row query tile, query head).
//   scores = (QK^T / sqrt(D))^2 * exp(logG_t - logG_s), causal, row-normalized
// Q: [T][QDIM] bf16, K: [T][KVDIM] bf16, Vt: [KVDIM][T] bf16, lG: [T][H] f32
// ---------------------------------------------------------------------------
__global__ void __launch_bounds__(32)
power_attn(const unsigned short* __restrict__ Qb,
           const unsigned short* __restrict__ Kb,
           const unsigned short* __restrict__ Vt,
           const float* __restrict__ lG,
           unsigned short* __restrict__ Yb) {
  __shared__ unsigned short smem[16 * 32];   // gated-score tile, A-operand staging
  const int lane = threadIdx.x;
  const int h  = lane >> 4;
  const int nn = lane & 15;
  const int t0 = blockIdx.x * 16;
  const int hq = blockIdx.y;
  const int hk = hq >> 1;                    // GQA ratio R=2
  const float scale = 0.08838834764831845f;  // 1/sqrt(128)

  // Q fragments for the whole tile: K = 128 -> 4 chunks of 32
  bf16x16 qa[4];
  const unsigned short* qrow = Qb + (size_t)(t0 + nn) * QDIM + hq * HEADD;
#pragma unroll
  for (int kc = 0; kc < 4; ++kc) qa[kc] = load_a_frag(qrow + kc * 32, h);

  float gt[8];
#pragma unroll
  for (int r = 0; r < 8; ++r) gt[r] = lG[(t0 + r + 8 * h) * NHEADS + hk];

  f32x8 yacc[8];
#pragma unroll
  for (int d = 0; d < 8; ++d) yacc[d] = (f32x8){};
  float rs[8];
#pragma unroll
  for (int r = 0; r < 8; ++r) rs[r] = 0.f;

  const int nblk = (t0 + 16 + 31) / 32;      // 32 keys per step
  for (int ib = 0; ib < nblk; ++ib) {
    const int sb = ib * 32;
    const int s0 = sb + nn;
    const int s1 = sb + 16 + nn;
    const int s0c = (s0 < T_DIM) ? s0 : (T_DIM - 1);
    const int s1c = (s1 < T_DIM) ? s1 : (T_DIM - 1);

    // QK^T for two 16-key tiles
    const unsigned short* k0p = Kb + (size_t)s0c * KVDIM + hk * HEADD + 16 * h;
    const unsigned short* k1p = Kb + (size_t)s1c * KVDIM + hk * HEADD + 16 * h;
    f32x8 acc0 = {}, acc1 = {};
#pragma unroll
    for (int kc = 0; kc < 4; ++kc) {
      acc0 = wmma_bf16(qa[kc], load_row16(k0p + kc * 32), acc0);
      acc1 = wmma_bf16(qa[kc], load_row16(k1p + kc * 32), acc1);
    }

    const float gs0 = lG[s0c * NHEADS + hk];
    const float gs1 = lG[s1c * NHEADS + hk];
#pragma unroll
    for (int r = 0; r < 8; ++r) {
      const int t = t0 + r + 8 * h;
      float a0 = acc0[r] * scale; a0 *= a0;
      float a1 = acc1[r] * scale; a1 *= a1;
      float v0 = (s0 <= t) ? a0 * __expf(gt[r] - gs0) : 0.f;
      float v1 = (s1 <= t) ? a1 * __expf(gt[r] - gs1) : 0.f;
      rs[r] += v0 + v1;
      smem[(r + 8 * h) * 32 + nn]      = f32_to_bf16_rne(v0);
      smem[(r + 8 * h) * 32 + 16 + nn] = f32_to_bf16_rne(v1);
    }
    __syncthreads();

    // A (16x32 gated scores) * V (32 x 128) -> accumulate y
    bf16x16 af = load_a_frag(smem + nn * 32, h);
    int sbase = sb + 16 * h;
    if (sbase > T_DIM - 16) sbase = T_DIM - 16;   // fully-masked tail only
#pragma unroll
    for (int dt = 0; dt < 8; ++dt) {
      const unsigned short* vp =
          Vt + (size_t)(hk * HEADD + dt * 16 + nn) * T_DIM + sbase;
      yacc[dt] = wmma_bf16(af, load_row16(vp), yacc[dt]);
    }
    __syncthreads();
  }

  // row-sum across the 16 lanes of each half (rows r + 8*h)
#pragma unroll
  for (int r = 0; r < 8; ++r) {
    float v = rs[r];
    v += __shfl_xor(v, 1); v += __shfl_xor(v, 2);
    v += __shfl_xor(v, 4); v += __shfl_xor(v, 8);
    rs[r] = v;
  }
#pragma unroll
  for (int r = 0; r < 8; ++r) {
    const float inv = 1.f / (rs[r] + 1e-6f);
    const int t = t0 + r + 8 * h;
#pragma unroll
    for (int dt = 0; dt < 8; ++dt) {
      Yb[(size_t)t * QDIM + hq * HEADD + dt * 16 + nn] =
          f32_to_bf16_rne(yacc[dt][r] * inv);
    }
  }
}

// ---------------------------------------------------------------------------
extern "C" void kernel_launch(void* const* d_in, const int* in_sizes, int n_in,
                              void* d_out, int out_size, void* d_ws, size_t ws_size,
                              hipStream_t stream) {
  (void)in_sizes; (void)n_in; (void)out_size; (void)ws_size;
  const float* X  = (const float*)d_in[0];
  const float* Wq = (const float*)d_in[1];
  const float* Wk = (const float*)d_in[2];
  const float* Wv = (const float*)d_in[3];
  const float* Wg = (const float*)d_in[4];
  const float* Wc = (const float*)d_in[5];
  const float* bc = (const float*)d_in[6];
  float* out = (float*)d_out;

  char* ws = (char*)d_ws;
  size_t off = 0;
  auto alloc = [&](size_t bytes) -> char* {
    char* p = ws + off;
    off += (bytes + 255) & ~(size_t)255;
    return p;
  };
  unsigned short* Xb  = (unsigned short*)alloc((size_t)T_DIM * C_DIM * 2);
  unsigned short* WqT = (unsigned short*)alloc((size_t)QDIM * C_DIM * 2);
  unsigned short* WkT = (unsigned short*)alloc((size_t)KVDIM * C_DIM * 2);
  unsigned short* WvT = (unsigned short*)alloc((size_t)KVDIM * C_DIM * 2);
  unsigned short* WcT = (unsigned short*)alloc((size_t)C_DIM * QDIM * 2);
  unsigned short* Qb  = (unsigned short*)alloc((size_t)T_DIM * QDIM * 2);
  unsigned short* Kb  = (unsigned short*)alloc((size_t)T_DIM * KVDIM * 2);
  unsigned short* Vt  = (unsigned short*)alloc((size_t)KVDIM * T_DIM * 2);
  unsigned short* Yb  = (unsigned short*)alloc((size_t)T_DIM * QDIM * 2);
  float* lgRaw = (float*)alloc((size_t)T_DIM * NHEADS * 4);
  float* lG    = (float*)alloc((size_t)T_DIM * NHEADS * 4);

  // bf16 conversions (weights transposed for contiguous B-operand fragments)
  conv_f32_to_bf16<<<(T_DIM * C_DIM + 255) / 256, 256, 0, stream>>>(X, Xb, T_DIM * C_DIM);
  conv_transpose_f32_to_bf16<<<(C_DIM * QDIM + 255) / 256, 256, 0, stream>>>(Wq, WqT, C_DIM, QDIM);
  conv_transpose_f32_to_bf16<<<(C_DIM * KVDIM + 255) / 256, 256, 0, stream>>>(Wk, WkT, C_DIM, KVDIM);
  conv_transpose_f32_to_bf16<<<(C_DIM * KVDIM + 255) / 256, 256, 0, stream>>>(Wv, WvT, C_DIM, KVDIM);
  conv_transpose_f32_to_bf16<<<(QDIM * C_DIM + 255) / 256, 256, 0, stream>>>(Wc, WcT, QDIM, C_DIM);

  // gating
  gate_logsig<<<T_DIM, 256, 0, stream>>>(X, Wg, lgRaw);
  gate_cumsum<<<1, NHEADS, 0, stream>>>(lgRaw, lG);

  // projections (bf16 WMMA, 128x128 block / 32x64 wave tiles)
  dim3 blk(256);
  gemm_bf16_bf16out<<<dim3(QDIM / 128, T_DIM / 128), blk, 0, stream>>>(
      Xb, WqT, Qb, C_DIM, C_DIM, C_DIM, QDIM, 0);
  gemm_bf16_bf16out<<<dim3(KVDIM / 128, T_DIM / 128), blk, 0, stream>>>(
      Xb, WkT, Kb, C_DIM, C_DIM, C_DIM, KVDIM, 0);
  gemm_bf16_bf16out<<<dim3(KVDIM / 128, T_DIM / 128), blk, 0, stream>>>(
      Xb, WvT, Vt, C_DIM, C_DIM, C_DIM, /*ldc=*/T_DIM, 1);  // V stored transposed

  // attention (WMMA for QK^T and A*V)
  power_attn<<<dim3(T_DIM / 16, HQ), 32, 0, stream>>>(Qb, Kb, Vt, lG, Yb);

  // output projection + bias (f32 out)
  gemm_bf16_f32out<<<dim3(C_DIM / 128, T_DIM / 128), blk, 0, stream>>>(
      Yb, WcT, bc, out, QDIM, QDIM, QDIM, C_DIM);
}
